// BidirectionalLSTM_6674379178403
// MI455X (gfx1250) — compile-verified, matrix operand order checked by reading, not run on hardware
//
#include <hip/hip_runtime.h>

// ---------------- problem sizes ----------------
#define TT   128
#define BB   128
#define NIN  512
#define NH   512
#define NOUT 512
#define G4   (4 * NH)      // 2048 gate width
#define MROWS (TT * BB)    // 16384

typedef __bf16 bf16x16 __attribute__((ext_vector_type(16)));
typedef float  f32x8   __attribute__((ext_vector_type(8)));

union FragB16 { uint4 q[2]; bf16x16 v; };

__device__ __forceinline__ unsigned short f32_to_bf16_bits(float f) {
    unsigned int u = __builtin_bit_cast(unsigned int, f);
    unsigned int r = u + 0x7FFFu + ((u >> 16) & 1u);   // round-to-nearest-even
    return (unsigned short)(r >> 16);
}

__device__ __forceinline__ float sigmoidf_(float x) {
    return 1.0f / (1.0f + __expf(-x));
}

// Load one 16x32 (rows x K) bf16 WMMA A/B fragment from row-major storage
// (K contiguous, row stride `ld` elements). Per the CDNA5 ISA 16-bit layout:
// lanes 0-15 hold K = {kBase+0..7, kBase+16..23}, lanes 16-31 hold
// K = {kBase+8..15, kBase+24..31}; i.e. two contiguous 16B runs per lane.
__device__ __forceinline__ bf16x16 load_frag(const unsigned short* base,
                                             int rowBase, int kBase, int ld) {
    int lane = threadIdx.x & 31;
    int half = lane >> 4;
    int r    = lane & 15;
    const unsigned short* p = base + (size_t)(rowBase + r) * ld + kBase + half * 8;
    FragB16 f;
    f.q[0] = *reinterpret_cast<const uint4*>(p);
    f.q[1] = *reinterpret_cast<const uint4*>(p + 16);
    return f.v;
}

__device__ __forceinline__ f32x8 wmma_bf16(bf16x16 a, bf16x16 b, f32x8 c) {
    return __builtin_amdgcn_wmma_f32_16x16x32_bf16(false, a, false, b,
                                                   (short)0, c, false, false);
}

// ---------------- f32 -> bf16 conversion (8 elements / thread) ----------------
__global__ void __launch_bounds__(256) cvt_bf16_kernel(const float* __restrict__ in,
                                                       unsigned int* __restrict__ out,
                                                       int n8) {
    int i = blockIdx.x * 256 + threadIdx.x;
    if (i < n8) {
        const float4* p = reinterpret_cast<const float4*>(in) + 2 * (size_t)i;
        float4 a = p[0], b = p[1];
        uint4 o;
        o.x = (unsigned)f32_to_bf16_bits(a.x) | ((unsigned)f32_to_bf16_bits(a.y) << 16);
        o.y = (unsigned)f32_to_bf16_bits(a.z) | ((unsigned)f32_to_bf16_bits(a.w) << 16);
        o.z = (unsigned)f32_to_bf16_bits(b.x) | ((unsigned)f32_to_bf16_bits(b.y) << 16);
        o.w = (unsigned)f32_to_bf16_bits(b.z) | ((unsigned)f32_to_bf16_bits(b.w) << 16);
        reinterpret_cast<uint4*>(out)[i] = o;
    }
}

// ---------------- input projection: xp = x @ W_ih^T + b_ih + b_hh ----------------
// M = 16384, N = 2048 (gates), K = 512. One wave owns a 16x64 output tile.
// Fully-unrolled k-loop, ping-pong prefetch (no register rotation copies).
__global__ void __launch_bounds__(256) ip_gemm_kernel(const unsigned short* __restrict__ xbf,
                                                      const unsigned short* __restrict__ wih,
                                                      const float* __restrict__ bih,
                                                      const float* __restrict__ bhh,
                                                      float* __restrict__ xp) {
    const int NT4 = G4 / 64;                              // 32 column tiles
    int wave = (blockIdx.x * blockDim.x + threadIdx.x) >> 5;
    int m0 = (wave / NT4) * 16;
    int n0 = (wave % NT4) * 64;

    f32x8 acc[4] = {};
    bf16x16 a0, a1, b0[4], b1[4];
    a0 = load_frag(xbf, m0, 0, NIN);
    #pragma unroll
    for (int j = 0; j < 4; ++j) b0[j] = load_frag(wih, n0 + j * 16, 0, NIN);

    #pragma unroll
    for (int kk = 0; kk < NIN / 64; ++kk) {               // 8 unrolled iterations
        int k = kk * 64;
        a1 = load_frag(xbf, m0, k + 32, NIN);
        #pragma unroll
        for (int j = 0; j < 4; ++j) b1[j] = load_frag(wih, n0 + j * 16, k + 32, NIN);
        #pragma unroll
        for (int j = 0; j < 4; ++j) acc[j] = wmma_bf16(a0, b0[j], acc[j]);
        if (kk + 1 < NIN / 64) {                          // compile-time guard
            a0 = load_frag(xbf, m0, k + 64, NIN);
            #pragma unroll
            for (int j = 0; j < 4; ++j) b0[j] = load_frag(wih, n0 + j * 16, k + 64, NIN);
        }
        #pragma unroll
        for (int j = 0; j < 4; ++j) acc[j] = wmma_bf16(a1, b1[j], acc[j]);
    }

    int lane = threadIdx.x & 31, half = lane >> 4, nl = lane & 15;
    #pragma unroll
    for (int j = 0; j < 4; ++j) {
        int n = n0 + j * 16 + nl;
        float bias = bih[n] + bhh[n];
        #pragma unroll
        for (int v = 0; v < 8; ++v) {
            int m = m0 + half * 8 + v;
            xp[(size_t)m * G4 + n] = acc[j][v] + bias;
        }
    }
}

// ---------------- one recurrent timestep, both directions ----------------
// gates = xp[t] + h_prev @ W_hh^T, then fused LSTM cell elementwise.
// One wave owns 16 batch rows x 16 hidden cols, accumulating all 4 gates.
__global__ void __launch_bounds__(256) lstm_step_kernel(const float* __restrict__ xpF,
                                                        const float* __restrict__ xpR,
                                                        const unsigned short* __restrict__ whhF,
                                                        const unsigned short* __restrict__ whhR,
                                                        unsigned short* __restrict__ hF,
                                                        unsigned short* __restrict__ hR,
                                                        float* __restrict__ cF,
                                                        float* __restrict__ cR,
                                                        int s) {
    int dir = blockIdx.y;
    const float* xp;  const unsigned short* whh;  unsigned short* h;  float* c;
    int t, tp;
    if (dir == 0) { xp = xpF; whh = whhF; h = hF; c = cF; t = s;          tp = s - 1;  }
    else          { xp = xpR; whh = whhR; h = hR; c = cR; t = TT - 1 - s; tp = TT - s; }

    const int NT = NH / 16;                                // 32 hidden tiles
    int wave = (blockIdx.x * blockDim.x + threadIdx.x) >> 5;
    int m0 = (wave / NT) * 16;                             // batch tile
    int n0 = (wave % NT) * 16;                             // hidden tile
    int lane = threadIdx.x & 31, half = lane >> 4, nl = lane & 15;

    // accumulators start from the precomputed input projection
    f32x8 acc[4];
    const float* xpt = xp + (size_t)t * BB * G4;
    #pragma unroll
    for (int q = 0; q < 4; ++q)
        #pragma unroll
        for (int v = 0; v < 8; ++v) {
            int m = m0 + half * 8 + v;
            acc[q][v] = xpt[(size_t)m * G4 + q * NH + n0 + nl];
        }

    if (s != 0) {   // wave-uniform branch; EXEC stays all-ones inside
        const unsigned short* hprev = h + (size_t)tp * BB * NH;
        bf16x16 a0, a1, b0[4], b1[4];
        a0 = load_frag(hprev, m0, 0, NH);
        #pragma unroll
        for (int q = 0; q < 4; ++q) b0[q] = load_frag(whh, q * NH + n0, 0, NH);

        #pragma unroll
        for (int kk = 0; kk < NH / 64; ++kk) {             // 8 unrolled iterations
            int k = kk * 64;
            a1 = load_frag(hprev, m0, k + 32, NH);
            #pragma unroll
            for (int q = 0; q < 4; ++q) b1[q] = load_frag(whh, q * NH + n0, k + 32, NH);
            #pragma unroll
            for (int q = 0; q < 4; ++q) acc[q] = wmma_bf16(a0, b0[q], acc[q]);
            if (kk + 1 < NH / 64) {
                a0 = load_frag(hprev, m0, k + 64, NH);
                #pragma unroll
                for (int q = 0; q < 4; ++q) b0[q] = load_frag(whh, q * NH + n0, k + 64, NH);
            }
            #pragma unroll
            for (int q = 0; q < 4; ++q) acc[q] = wmma_bf16(a1, b1[q], acc[q]);
        }
    }

    // fused LSTM cell (PyTorch gate order i, f, g, o)
    unsigned short* ht = h + (size_t)t * BB * NH;
    #pragma unroll
    for (int v = 0; v < 8; ++v) {
        int m  = m0 + half * 8 + v;
        int hn = n0 + nl;
        float ig = sigmoidf_(acc[0][v]);
        float fg = sigmoidf_(acc[1][v]);
        float gg = tanhf(acc[2][v]);
        float og = sigmoidf_(acc[3][v]);
        float cp = (s == 0) ? 0.0f : c[(size_t)m * NH + hn];
        float cn = fg * cp + ig * gg;
        c[(size_t)m * NH + hn] = cn;
        ht[(size_t)m * NH + hn] = f32_to_bf16_bits(og * tanhf(cn));
    }
}

// ---------------- output projection: out = [h_f|h_r] @ W_emb^T + b_emb ----------------
// M = 16384, N = 512, K = 1024 (split-K over the two h buffers).
__global__ void __launch_bounds__(256) out_gemm_kernel(const unsigned short* __restrict__ hF,
                                                       const unsigned short* __restrict__ hR,
                                                       const unsigned short* __restrict__ wemb,
                                                       const float* __restrict__ bemb,
                                                       float* __restrict__ out) {
    const int NT4 = NOUT / 64;                             // 8 column tiles
    int wave = (blockIdx.x * blockDim.x + threadIdx.x) >> 5;
    int m0 = (wave / NT4) * 16;
    int n0 = (wave % NT4) * 64;

    // A-fragment selector over the concatenated [hF | hR] K range (constant-folds)
    auto loadA = [&](int k) -> bf16x16 {
        const unsigned short* hbuf = (k < NH) ? hF : hR;
        return load_frag(hbuf, m0, (k < NH) ? k : k - NH, NH);
    };

    f32x8 acc[4] = {};
    bf16x16 a0, a1, b0[4], b1[4];
    a0 = loadA(0);
    #pragma unroll
    for (int j = 0; j < 4; ++j) b0[j] = load_frag(wemb, n0 + j * 16, 0, 2 * NH);

    #pragma unroll
    for (int kk = 0; kk < 2 * NH / 64; ++kk) {             // 16 unrolled iterations
        int k = kk * 64;
        a1 = loadA(k + 32);
        #pragma unroll
        for (int j = 0; j < 4; ++j) b1[j] = load_frag(wemb, n0 + j * 16, k + 32, 2 * NH);
        #pragma unroll
        for (int j = 0; j < 4; ++j) acc[j] = wmma_bf16(a0, b0[j], acc[j]);
        if (kk + 1 < 2 * NH / 64) {
            a0 = loadA(k + 64);
            #pragma unroll
            for (int j = 0; j < 4; ++j) b0[j] = load_frag(wemb, n0 + j * 16, k + 64, 2 * NH);
        }
        #pragma unroll
        for (int j = 0; j < 4; ++j) acc[j] = wmma_bf16(a1, b1[j], acc[j]);
    }

    int lane = threadIdx.x & 31, half = lane >> 4, nl = lane & 15;
    #pragma unroll
    for (int j = 0; j < 4; ++j) {
        int n = n0 + j * 16 + nl;
        float bias = bemb[n];
        #pragma unroll
        for (int v = 0; v < 8; ++v) {
            int m = m0 + half * 8 + v;
            out[(size_t)m * NOUT + n] = acc[j][v] + bias;
        }
    }
}

extern "C" void kernel_launch(void* const* d_in, const int* in_sizes, int n_in,
                              void* d_out, int out_size, void* d_ws, size_t ws_size,
                              hipStream_t stream) {
    const float* x      = (const float*)d_in[0];
    const float* W_ih_f = (const float*)d_in[1];
    const float* W_hh_f = (const float*)d_in[2];
    const float* b_ih_f = (const float*)d_in[3];
    const float* b_hh_f = (const float*)d_in[4];
    const float* W_ih_r = (const float*)d_in[5];
    const float* W_hh_r = (const float*)d_in[6];
    const float* b_ih_r = (const float*)d_in[7];
    const float* b_hh_r = (const float*)d_in[8];
    const float* W_emb  = (const float*)d_in[9];
    const float* b_emb  = (const float*)d_in[10];
    float* out = (float*)d_out;

    // ---- workspace layout (all offsets 256B aligned) ----
    char* ws = (char*)d_ws;
    size_t off = 0;
    auto take = [&](size_t bytes) { char* p = ws + off; off = (off + bytes + 255) & ~(size_t)255; return p; };

    unsigned short* xbf  = (unsigned short*)take((size_t)MROWS * NIN * 2);      // 16 MB
    unsigned short* wihF = (unsigned short*)take((size_t)G4 * NIN * 2);         // 2 MB
    unsigned short* whhF = (unsigned short*)take((size_t)G4 * NH  * 2);         // 2 MB
    unsigned short* wihR = (unsigned short*)take((size_t)G4 * NIN * 2);
    unsigned short* whhR = (unsigned short*)take((size_t)G4 * NH  * 2);
    unsigned short* wemb = (unsigned short*)take((size_t)NOUT * 2 * NH * 2);    // 1 MB
    float* xpF = (float*)take((size_t)MROWS * G4 * 4);                          // 128 MB
    float* xpR = (float*)take((size_t)MROWS * G4 * 4);                          // 128 MB
    unsigned short* hF = (unsigned short*)take((size_t)MROWS * NH * 2);         // 16 MB
    unsigned short* hR = (unsigned short*)take((size_t)MROWS * NH * 2);
    float* cF = (float*)take((size_t)BB * NH * 4);                              // 256 KB
    float* cR = (float*)take((size_t)BB * NH * 4);
    (void)ws_size; (void)in_sizes; (void)n_in; (void)out_size;

    // ---- 1) convert activations + weights to bf16 (8 elems / thread) ----
    auto cvt = [&](const float* src, unsigned short* dst, int n) {
        int n8 = n / 8;
        cvt_bf16_kernel<<<(n8 + 255) / 256, 256, 0, stream>>>(src, (unsigned int*)dst, n8);
    };
    cvt(x,      xbf,  MROWS * NIN);
    cvt(W_ih_f, wihF, G4 * NIN);
    cvt(W_hh_f, whhF, G4 * NH);
    cvt(W_ih_r, wihR, G4 * NIN);
    cvt(W_hh_r, whhR, G4 * NH);
    cvt(W_emb,  wemb, NOUT * 2 * NH);

    // ---- 2) input projections (off the serial path) ----
    {
        int waves = (MROWS / 16) * (G4 / 64);          // 32768 waves
        int blocks = waves * 32 / 256;                 // 4096 blocks
        ip_gemm_kernel<<<blocks, 256, 0, stream>>>(xbf, wihF, b_ih_f, b_hh_f, xpF);
        ip_gemm_kernel<<<blocks, 256, 0, stream>>>(xbf, wihR, b_ih_r, b_hh_r, xpR);
    }

    // ---- 3) T sequential recurrent steps, both directions per launch ----
    {
        int wavesPerDir = (BB / 16) * (NH / 16);       // 256 waves
        dim3 grid(wavesPerDir * 32 / 256, 2);          // (32, 2)
        for (int s = 0; s < TT; ++s)
            lstm_step_kernel<<<grid, 256, 0, stream>>>(xpF, xpR, whhF, whhR,
                                                       hF, hR, cF, cR, s);
    }

    // ---- 4) output projection ----
    {
        int waves = (MROWS / 16) * (NOUT / 64);        // 8192 waves
        int blocks = waves * 32 / 256;                 // 1024 blocks
        out_gemm_kernel<<<blocks, 256, 0, stream>>>(hF, hR, wemb, b_emb, out);
    }
}
